// CrossAttention_26276609917283
// MI455X (gfx1250) — matrix operand order repaired and checked by reading, not compiled, not run on hardware
//
#include <hip/hip_runtime.h>
#include <hip/hip_bf16.h>
#include <math.h>

// Problem constants (fixed by the reference)
#define B_    2
#define NQ_   2048
#define NKV_  4096
#define D_    1024
#define H_    16
#define DH_   64
#define HID_  8192
#define HIDH_ 4096

typedef _Float16       half8   __attribute__((ext_vector_type(8)));
typedef _Float16       v16h    __attribute__((ext_vector_type(16)));
typedef float          v8f     __attribute__((ext_vector_type(8)));
typedef unsigned short ushort8 __attribute__((ext_vector_type(8)));
typedef int            v4i_gcc __attribute__((vector_size(16)));

union H8U8 { half8 h; ushort8 u; };

// ---- async global->LDS staging (CDNA5 GLOBAL_LOAD_ASYNC_TO_LDS_B128), with fallback ----
#if defined(__has_builtin)
#  if __has_builtin(__builtin_amdgcn_global_load_async_to_lds_b128) && \
      __has_builtin(__builtin_amdgcn_s_wait_asynccnt)
#    define USE_ASYNC_LDS 1
#  endif
#endif
#ifndef USE_ASYNC_LDS
#  define USE_ASYNC_LDS 0
#endif

__device__ __forceinline__ void cp16(_Float16* l, const _Float16* g) {
#if USE_ASYNC_LDS
  // Builtin signature (from clang diagnostic): (v4i AS1*, v4i AS3*, imm offset, imm cpol)
  __builtin_amdgcn_global_load_async_to_lds_b128(
      (__attribute__((address_space(1))) v4i_gcc*)(_Float16*)g,
      (__attribute__((address_space(3))) v4i_gcc*)l, 0, 0);
#else
  *reinterpret_cast<uint4*>(l) = *reinterpret_cast<const uint4*>(g);
#endif
}
__device__ __forceinline__ void wait_async_all() {
#if USE_ASYNC_LDS
  __builtin_amdgcn_s_wait_asynccnt(0);
#endif
}

// Build a 16-half WMMA operand from two contiguous 8-half (16B) LDS chunks.
__device__ __forceinline__ v16h frag_pair(const _Float16* p0, const _Float16* p1) {
  half8 lo = *reinterpret_cast<const half8*>(p0);
  half8 hi = *reinterpret_cast<const half8*>(p1);
  v16h r;
#pragma unroll
  for (int i = 0; i < 8; ++i) { r[i] = lo[i]; r[i + 8] = hi[i]; }
  return r;
}

// ---------------- LayerNorm (fp32 in -> f16 out) ----------------
__global__ __launch_bounds__(256)
void ln_to_f16(const float* __restrict__ x, const float* __restrict__ scale,
               const float* __restrict__ bias, _Float16* __restrict__ y, int D) {
  const int row = blockIdx.x;
  const int tid = threadIdx.x;
  const float* xr = x + (size_t)row * D;
  float s = 0.f, s2 = 0.f;
  for (int i = tid; i < D; i += 256) { float v = xr[i]; s += v; s2 += v * v; }
  __shared__ float r1[256], r2[256];
  r1[tid] = s; r2[tid] = s2; __syncthreads();
  for (int off = 128; off > 0; off >>= 1) {
    if (tid < off) { r1[tid] += r1[tid + off]; r2[tid] += r2[tid + off]; }
    __syncthreads();
  }
  const float mu   = r1[0] / D;
  const float var  = r2[0] / D - mu * mu;
  const float rstd = rsqrtf(var + 1e-6f);
  _Float16* yr = y + (size_t)row * D;
  for (int i = tid; i < D; i += 256)
    yr[i] = (_Float16)((xr[i] - mu) * rstd * scale[i] + bias[i]);
}

// -------- Transpose + convert weights: src[R][C] f32 -> dst[C][R] f16 --------
__global__ __launch_bounds__(256)
void transpose_convert_f16(const float* __restrict__ src, _Float16* __restrict__ dst,
                           int R, int C) {
  __shared__ float tile[32][33];
  const int c0 = blockIdx.x * 32, r0 = blockIdx.y * 32;
  const int tx = threadIdx.x, ty = threadIdx.y; // (32, 8)
#pragma unroll
  for (int i = 0; i < 4; ++i)
    tile[ty + 8 * i][tx] = src[(size_t)(r0 + ty + 8 * i) * C + c0 + tx];
  __syncthreads();
#pragma unroll
  for (int i = 0; i < 4; ++i)
    dst[(size_t)(c0 + ty + 8 * i) * R + r0 + tx] = (_Float16)tile[tx][ty + 8 * i];
}

// ------------- Generic WMMA GEMM: out = A[M][K] * Bt[N][K]^T + bias (+res) -------------
// Block = 256 threads = 8 waves; block tile 128(M) x 64(N); wave tile 32x32.
// K tile = 64 (2 WMMA k-substeps), double-buffered LDS, async global->LDS staging.
template <typename OutT, bool RES>
__global__ __launch_bounds__(256)
void gemm_f16_wmma(const _Float16* __restrict__ A, const _Float16* __restrict__ Bt,
                   const float* __restrict__ bias, const float* __restrict__ res,
                   OutT* __restrict__ out, int M, int N, int K) {
  __shared__ __align__(16) _Float16 As[2][128 * 72]; // 128 x 64, row stride 72
  __shared__ __align__(16) _Float16 Bs[2][64 * 72];  // 64 x 64 (n-major), row stride 72

  const int tid  = threadIdx.x;
  const int lane = tid & 31, w = tid >> 5;
  const int g = lane >> 4, ln16 = lane & 15;
  const int mw = (w & 3) * 32, nw = (w >> 2) * 32;
  const int rowBase = blockIdx.y * 128;
  const int colBase = blockIdx.x * 64;

  const int aRow = tid >> 1;        // 0..127
  const int aCol = (tid & 1) * 32;  // 0 / 32 (4 x 16B chunks)
  const int bRow = tid >> 2;        // 0..63
  const int bCol = (tid & 3) * 16;  // (2 x 16B chunks)

  v8f acc[2][2] = {};

  auto stage = [&](int buf, int k0) {
    const _Float16* ga = A + (size_t)(rowBase + aRow) * K + k0 + aCol;
    _Float16* la = &As[buf][aRow * 72 + aCol];
#pragma unroll
    for (int c = 0; c < 4; ++c) cp16(la + c * 8, ga + c * 8);
    const _Float16* gb = Bt + (size_t)(colBase + bRow) * K + k0 + bCol;
    _Float16* lb = &Bs[buf][bRow * 72 + bCol];
#pragma unroll
    for (int c = 0; c < 2; ++c) cp16(lb + c * 8, gb + c * 8);
  };

  auto compute = [&](int buf) {
#pragma unroll
    for (int kk = 0; kk < 64; kk += 32) {
      v16h af[2], bf[2];
#pragma unroll
      for (int mi = 0; mi < 2; ++mi) {
        const _Float16* ap = &As[buf][(mw + mi * 16 + ln16) * 72 + kk];
        af[mi] = frag_pair(ap + g * 8, ap + 16 + g * 8);
      }
#pragma unroll
      for (int ni = 0; ni < 2; ++ni) {
        const _Float16* bp = &Bs[buf][(nw + ni * 16 + ln16) * 72 + kk + g * 8];
        bf[ni] = frag_pair(bp, bp + 16);
      }
#pragma unroll
      for (int mi = 0; mi < 2; ++mi)
#pragma unroll
        for (int ni = 0; ni < 2; ++ni)
          acc[mi][ni] = __builtin_amdgcn_wmma_f32_16x16x32_f16(
              false, af[mi], false, bf[ni], (short)0, acc[mi][ni], false, false);
    }
  };

  stage(0, 0);
  wait_async_all();
  __syncthreads();
  int cur = 0;
  const int nk = K >> 6;
  for (int kt = 0; kt < nk; ++kt) {
    if (kt + 1 < nk) stage(cur ^ 1, (kt + 1) * 64);
    compute(cur);
    wait_async_all();
    __syncthreads();
    cur ^= 1;
  }

#pragma unroll
  for (int mi = 0; mi < 2; ++mi)
#pragma unroll
    for (int ni = 0; ni < 2; ++ni) {
      const int col = colBase + nw + ni * 16 + ln16;
      const float bv = bias[col];
#pragma unroll
      for (int r = 0; r < 8; ++r) {
        const int row = rowBase + mw + mi * 16 + r + 8 * g;
        float v = acc[mi][ni][r] + bv;
        if (RES) v += res[(size_t)row * N + col];
        out[(size_t)row * N + col] = (OutT)v;
      }
    }
}

// ---------------- Flash attention: 64-query tile per block, per (b, h) ----------------
__global__ __launch_bounds__(256)
void attention_wmma(const _Float16* __restrict__ Q, const _Float16* __restrict__ Km,
                    const _Float16* __restrict__ Vm, _Float16* __restrict__ O) {
  __shared__ __align__(16) _Float16 Qs[64 * 72];   // [q][d]
  __shared__ __align__(16) _Float16 Ks[64 * 72];   // [kv][d]  (n-major for S)
  __shared__ __align__(16) _Float16 Ps[64 * 72];   // [q][kv] probabilities (A for P*V)
  __shared__ __align__(16) _Float16 Vts[64 * 72];  // [dh][kv] (n-major for P*V)
  __shared__ float Ss[64 * 68];
  __shared__ float mst[64], lst[64], ast[64];

  const int tid  = threadIdx.x;
  const int lane = tid & 31, w = tid >> 5;
  const int g = lane >> 4, ln16 = lane & 15;
  const int b = blockIdx.z, h = blockIdx.y, q0 = blockIdx.x * 64;

  const int lrow = tid >> 2;        // 0..63
  const int lcol = (tid & 3) * 16;  // 0/16/32/48
  const int vr = tid & 31;          // kv row pair (0..31)
  const int vc = (tid >> 5) * 8;    // dh chunk (0..56)

  { // load Q tile (stays resident) via async path
    const _Float16* gq = Q + (((size_t)b * NQ_ + q0 + lrow) * H_ + h) * DH_ + lcol;
    _Float16* lq = &Qs[lrow * 72 + lcol];
    cp16(lq, gq);
    cp16(lq + 8, gq + 8);
  }
  if (tid < 64) { mst[tid] = -INFINITY; lst[tid] = 0.f; }

  const int mi  = w & 3;          // q subtile
  const int ni0 = (w >> 2) * 2;   // first dh subtile
  v8f oc[2] = {};

  for (int kv0 = 0; kv0 < NKV_; kv0 += 64) {
    { // K tile row-major (async); V tile transposed into [dh][kv] (b32-packed DS stores)
      const _Float16* gk = Km + (((size_t)b * NKV_ + kv0 + lrow) * H_ + h) * DH_ + lcol;
      _Float16* lk = &Ks[lrow * 72 + lcol];
      cp16(lk, gk);
      cp16(lk + 8, gk + 8);

      const _Float16* gv = Vm + (((size_t)b * NKV_ + kv0 + 2 * vr) * H_ + h) * DH_ + vc;
      H8U8 u0, u1;
      u0.h = *reinterpret_cast<const half8*>(gv);
      u1.h = *reinterpret_cast<const half8*>(gv + (size_t)H_ * DH_);
#pragma unroll
      for (int i = 0; i < 8; ++i) {
        const unsigned wv = (unsigned)u0.u[i] | ((unsigned)u1.u[i] << 16);
        *reinterpret_cast<unsigned*>(&Vts[(vc + i) * 72 + 2 * vr]) = wv;
      }
    }
    wait_async_all();
    __syncthreads();

    // S = (Q K^T) * 1/sqrt(DH); each wave computes 2 of 16 subtiles
#pragma unroll
    for (int t = 0; t < 2; ++t) {
      const int nj = ni0 + t;
      v8f s = {};
#pragma unroll
      for (int kk = 0; kk < 64; kk += 32) {
        const _Float16* ap = &Qs[(mi * 16 + ln16) * 72 + kk];
        v16h af = frag_pair(ap + g * 8, ap + 16 + g * 8);
        const _Float16* bp = &Ks[(nj * 16 + ln16) * 72 + kk + g * 8];
        v16h bf = frag_pair(bp, bp + 16);
        s = __builtin_amdgcn_wmma_f32_16x16x32_f16(false, af, false, bf, (short)0, s, false, false);
      }
#pragma unroll
      for (int r = 0; r < 8; ++r)
        Ss[(mi * 16 + r + 8 * g) * 68 + nj * 16 + ln16] = s[r] * 0.125f;
    }
    __syncthreads();

    // Online softmax stats (one thread per query row)
    if (tid < 64) {
      const float mo = mst[tid];
      float mn = mo;
      const float* sr = &Ss[tid * 68];
#pragma unroll 8
      for (int j = 0; j < 64; ++j) mn = fmaxf(mn, sr[j]);
      const float al = __expf(mo - mn);
      float sum = 0.f;
      _Float16* pr = &Ps[tid * 72];
#pragma unroll 8
      for (int j = 0; j < 64; ++j) { float p = __expf(sr[j] - mn); sum += p; pr[j] = (_Float16)p; }
      mst[tid] = mn; lst[tid] = lst[tid] * al + sum; ast[tid] = al;
    }
    __syncthreads();

    // Rescale running O, then O += P * V
#pragma unroll
    for (int t = 0; t < 2; ++t)
#pragma unroll
      for (int r = 0; r < 8; ++r) oc[t][r] *= ast[mi * 16 + r + 8 * g];
#pragma unroll
    for (int t = 0; t < 2; ++t) {
      const int ni = ni0 + t;
#pragma unroll
      for (int kk = 0; kk < 64; kk += 32) {
        const _Float16* ap = &Ps[(mi * 16 + ln16) * 72 + kk];
        v16h af = frag_pair(ap + g * 8, ap + 16 + g * 8);
        const _Float16* bp = &Vts[(ni * 16 + ln16) * 72 + kk + g * 8];
        v16h bf = frag_pair(bp, bp + 16);
        oc[t] = __builtin_amdgcn_wmma_f32_16x16x32_f16(false, af, false, bf, (short)0, oc[t], false, false);
      }
    }
    __syncthreads();
  }

  // Normalize by l and write O (f16)
#pragma unroll
  for (int t = 0; t < 2; ++t) {
    const int dh = (ni0 + t) * 16 + ln16;
#pragma unroll
    for (int r = 0; r < 8; ++r) {
      const int qr = mi * 16 + r + 8 * g;
      const float linv = 1.f / lst[qr];
      O[(((size_t)b * NQ_ + q0 + qr) * H_ + h) * DH_ + dh] = (_Float16)(oc[t][r] * linv);
    }
  }
}

// ---------------- GEGLU: hg[m][j] = h[m][j] * gelu_tanh(h[m][4096+j]) ----------------
__global__ __launch_bounds__(256)
void geglu_kernel(const _Float16* __restrict__ h1, _Float16* __restrict__ hg) {
  const size_t idx = (size_t)blockIdx.x * 256 + threadIdx.x;
  if (idx >= (size_t)B_ * NQ_ * HIDH_) return;
  const size_t m = idx / HIDH_, j = idx % HIDH_;
  const float a  = (float)h1[m * HID_ + j];
  const float gg = (float)h1[m * HID_ + HIDH_ + j];
  const float t  = tanhf(0.7978845608028654f * (gg + 0.044715f * gg * gg * gg));
  hg[idx] = (_Float16)(a * 0.5f * gg * (1.f + t));
}

// ---------------- Host orchestration ----------------
extern "C" void kernel_launch(void* const* d_in, const int* in_sizes, int n_in,
                              void* d_out, int out_size, void* d_ws, size_t ws_size,
                              hipStream_t stream) {
  const float* inputs_q  = (const float*)d_in[0];
  const float* inputs_kv = (const float*)d_in[1];
  const float* ln_q_s    = (const float*)d_in[2];
  const float* ln_q_b    = (const float*)d_in[3];
  const float* ln_kv_s   = (const float*)d_in[4];
  const float* ln_kv_b   = (const float*)d_in[5];
  const float* Wq = (const float*)d_in[6];
  const float* bq = (const float*)d_in[7];
  const float* Wk = (const float*)d_in[8];
  const float* bk = (const float*)d_in[9];
  const float* Wv = (const float*)d_in[10];
  const float* bv = (const float*)d_in[11];
  const float* Wo = (const float*)d_in[12];
  const float* bo = (const float*)d_in[13];
  const float* ln2_s = (const float*)d_in[14];
  const float* ln2_b = (const float*)d_in[15];
  const float* W1 = (const float*)d_in[16];
  const float* b1 = (const float*)d_in[17];
  const float* W2 = (const float*)d_in[18];
  const float* b2 = (const float*)d_in[19];
  float* out = (float*)d_out;

  // Workspace carve-up (~224 MB total, 256B aligned slices)
  char* ws = (char*)d_ws;
  size_t off = 0;
  auto alloc = [&](size_t bytes) -> char* {
    char* p = ws + off;
    off += (bytes + 255) & ~((size_t)255);
    return p;
  };
  const int MQ = B_ * NQ_;   // 4096 query rows
  const int MK = B_ * NKV_;  // 8192 kv rows

  _Float16* WqT = (_Float16*)alloc((size_t)D_ * D_ * 2);
  _Float16* WkT = (_Float16*)alloc((size_t)D_ * D_ * 2);
  _Float16* WvT = (_Float16*)alloc((size_t)D_ * D_ * 2);
  _Float16* WoT = (_Float16*)alloc((size_t)D_ * D_ * 2);
  _Float16* W1T = (_Float16*)alloc((size_t)HID_ * D_ * 2);
  _Float16* W2T = (_Float16*)alloc((size_t)D_ * HIDH_ * 2);
  _Float16* lnq  = (_Float16*)alloc((size_t)MQ * D_ * 2);
  _Float16* lnkv = (_Float16*)alloc((size_t)MK * D_ * 2);
  _Float16* qf = (_Float16*)alloc((size_t)MQ * D_ * 2);
  _Float16* kf = (_Float16*)alloc((size_t)MK * D_ * 2);
  _Float16* vf = (_Float16*)alloc((size_t)MK * D_ * 2);
  _Float16* of = (_Float16*)alloc((size_t)MQ * D_ * 2);
  float*    xr = (float*)   alloc((size_t)MQ * D_ * 4);
  _Float16* ln2f = (_Float16*)alloc((size_t)MQ * D_ * 2);
  _Float16* h1 = (_Float16*)alloc((size_t)MQ * HID_ * 2);
  _Float16* hg = (_Float16*)alloc((size_t)MQ * HIDH_ * 2);

  const dim3 tb(32, 8);
  // Weight transposes: src[R][C] -> dst[C][R]
  transpose_convert_f16<<<dim3(D_ / 32, D_ / 32), tb, 0, stream>>>(Wq, WqT, D_, D_);
  transpose_convert_f16<<<dim3(D_ / 32, D_ / 32), tb, 0, stream>>>(Wk, WkT, D_, D_);
  transpose_convert_f16<<<dim3(D_ / 32, D_ / 32), tb, 0, stream>>>(Wv, WvT, D_, D_);
  transpose_convert_f16<<<dim3(D_ / 32, D_ / 32), tb, 0, stream>>>(Wo, WoT, D_, D_);
  transpose_convert_f16<<<dim3(HID_ / 32, D_ / 32), tb, 0, stream>>>(W1, W1T, D_, HID_);
  transpose_convert_f16<<<dim3(D_ / 32, HIDH_ / 32), tb, 0, stream>>>(W2, W2T, HIDH_, D_);

  // LayerNorms -> f16
  ln_to_f16<<<MQ, 256, 0, stream>>>(inputs_q, ln_q_s, ln_q_b, lnq, D_);
  ln_to_f16<<<MK, 256, 0, stream>>>(inputs_kv, ln_kv_s, ln_kv_b, lnkv, D_);

  // Q/K/V projections (f16 out)
  gemm_f16_wmma<_Float16, false><<<dim3(D_ / 64, MQ / 128), 256, 0, stream>>>(
      lnq, WqT, bq, nullptr, qf, MQ, D_, D_);
  gemm_f16_wmma<_Float16, false><<<dim3(D_ / 64, MK / 128), 256, 0, stream>>>(
      lnkv, WkT, bk, nullptr, kf, MK, D_, D_);
  gemm_f16_wmma<_Float16, false><<<dim3(D_ / 64, MK / 128), 256, 0, stream>>>(
      lnkv, WvT, bv, nullptr, vf, MK, D_, D_);

  // Attention
  attention_wmma<<<dim3(NQ_ / 64, H_, B_), 256, 0, stream>>>(qf, kf, vf, of);

  // Output projection + query residual (f32)
  gemm_f16_wmma<float, true><<<dim3(D_ / 64, MQ / 128), 256, 0, stream>>>(
      of, WoT, bo, inputs_q, xr, MQ, D_, D_);

  // MLP: LN2 -> W1 -> GEGLU -> W2 (+x residual) into d_out
  ln_to_f16<<<MQ, 256, 0, stream>>>(xr, ln2_s, ln2_b, ln2f, D_);
  gemm_f16_wmma<_Float16, false><<<dim3(HID_ / 64, MQ / 128), 256, 0, stream>>>(
      ln2f, W1T, b1, nullptr, h1, MQ, HID_, D_);
  const size_t ng = (size_t)MQ * HIDH_;
  geglu_kernel<<<(unsigned)((ng + 255) / 256), 256, 0, stream>>>(h1, hg);
  gemm_f16_wmma<float, true><<<dim3(D_ / 64, MQ / 128), 256, 0, stream>>>(
      hg, W2T, b2, xr, out, MQ, D_, HIDH_);
}
